// BaseModel_38233798869553
// MI455X (gfx1250) — matrix operand order repaired and checked by reading, not compiled, not run on hardware
//
#include <hip/hip_runtime.h>
#include <hip/hip_bf16.h>

typedef __attribute__((ext_vector_type(16))) _Float16 v16h;
typedef __attribute__((ext_vector_type(8)))  _Float16 v8h;
typedef __attribute__((ext_vector_type(8)))  float    v8f;

union V16 { v16h v; v8h h[2]; };

#define TILE     32
#define NTHREADS 256
#define NWAVES   8

// f16 workspace layout (element offsets); all N padded to multiples of 64 rows
// (except FC2) and K padded to multiples of 32.
#define OFF_WC   0            // [256][64]   rows 0-99 = w3 tap1, 100-199 = w5 tap2
#define OFF_W11  16384        // [512][768]  (500x750 zero padded)
#define OFF_W12  409600       // [512][1024] (500x1000 zero padded)
#define OFF_FC1  933888       // [1024][512] (1024x500 zero padded)
#define OFF_FC2  1458176      // [16][1024]  (8x1024 zero padded)
#define WS_TOTAL 1474560      // ~2.95 MB

// ---------------------------------------------------------------------------
// Prep: repack weights fp32 -> fp16 (zero padded) so WMMA B-fragments are two
// contiguous global_load_b128 per lane. Stateless, runs every launch.
// ---------------------------------------------------------------------------
__global__ __launch_bounds__(256) void prep_weights(
    const float* __restrict__ w3, const float* __restrict__ w5,
    const float* __restrict__ w11, const float* __restrict__ w12,
    const float* __restrict__ fc1w, const float* __restrict__ fc2w,
    _Float16* __restrict__ ws)
{
    int i = blockIdx.x * 256 + threadIdx.x;
    if (i >= WS_TOTAL) return;
    float v = 0.0f;
    if (i < OFF_W11) {                       // Wc [256][64]
        int r = i >> 6, c = i & 63;
        if (c < 50) {
            if (r < 100)       v = w3[(r * 50 + c) * 3 + 1];        // K=3 center tap
            else if (r < 200)  v = w5[((r - 100) * 50 + c) * 5 + 2];// K=5 center tap
        }
    } else if (i < OFF_W12) {                // W11 [512][768]
        int j = i - OFF_W11; int r = j / 768, c = j % 768;
        if (r < 500 && c < 750) v = w11[r * 750 + c];
    } else if (i < OFF_FC1) {                // W12 [512][1024]
        int j = i - OFF_W12; int r = j >> 10, c = j & 1023;
        if (r < 500 && c < 1000) v = w12[r * 1000 + c];
    } else if (i < OFF_FC2) {                // FC1 [1024][512]
        int j = i - OFF_FC1; int r = j >> 9, c = j & 511;
        if (c < 500) v = fc1w[r * 500 + c];
    } else {                                 // FC2 [16][1024]
        int j = i - OFF_FC2; int r = j >> 10, c = j & 1023;
        if (r < 8) v = fc2w[r * 1024 + c];
    }
    ws[i] = (_Float16)v;
}

// ---------------------------------------------------------------------------
// Fast activation helpers (v_exp_f32 + v_rcp_f32, no fp32 div sequences)
// ---------------------------------------------------------------------------
__device__ __forceinline__ float fast_sigmoid(float v) {
    return __builtin_amdgcn_rcpf(1.0f + __expf(-v));
}
__device__ __forceinline__ float fast_tanh(float v) {
    float e = __expf(2.0f * v);
    return 1.0f - 2.0f * __builtin_amdgcn_rcpf(e + 1.0f);   // robust at +/-inf
}

// Scalar-hidden GRU scan: lane = batch row within the tile (wave32).
__device__ void gru_scan(const _Float16* in, int inStride, int T,
                         const float* __restrict__ p,   // [4][3] wi, wh, bi, bh
                         _Float16* o, int oStride, int oOff, int lane)
{
    const float wi_r = p[0],  wi_z = p[1],  wi_n = p[2];
    const float wh_r = p[3],  wh_z = p[4],  wh_n = p[5];
    const float bi_r = p[6],  bi_z = p[7],  bi_n = p[8];
    const float bh_r = p[9],  bh_z = p[10], bh_n = p[11];
    const _Float16* ip = in + lane * inStride;
    _Float16* op = o + lane * oStride + oOff;
    float h = 0.0f;
    for (int t = 0; t < T; ++t) {
        float xt = (float)ip[t];
        float r = fast_sigmoid(fmaf(wi_r, xt, bi_r) + fmaf(wh_r, h, bh_r));
        float z = fast_sigmoid(fmaf(wi_z, xt, bi_z) + fmaf(wh_z, h, bh_z));
        float n = fast_tanh(fmaf(wi_n, xt, bi_n) + r * fmaf(wh_n, h, bh_n));
        h = fmaf(z, h - n, n);               // (1-z)*n + z*h
        op[t] = (_Float16)h;
    }
}

// ---------------------------------------------------------------------------
// WMMA GEMM over a 32-row batch tile, NT output-tile blocking per wave.
//   A: LDS activations [32][actStride] f16 (K mult of 32, zero padded)
//   B: global repacked weights [out][wStride] f16 (row = output channel)
// A-frag (contiguous k, same order as B): lanes 0-15 hold K 0-15 of row M=lane,
// lanes 16-31 hold K 16-31 -> two ds_load_b128.
// B-frag: lane holds 16 contiguous K of column n0+(lane&15); half-wave picks
// K 0-15 vs 16-31 -> two global_load_b128. Double-buffered over k.
// ---------------------------------------------------------------------------
template<int NT, bool RELU, bool TO_LDS>
__device__ __forceinline__ void gemm(
    const _Float16* act, int actStride, int K,
    const _Float16* __restrict__ W, int wStride, int nTiles,
    int lane, int wave,
    _Float16* dst, int dstStride, int dstColOff,
    const float* __restrict__ bias, int nMax,
    float* __restrict__ gOut, int b0)
{
    const int l15  = lane & 15;
    const int half = lane >> 4;
    const int items = 2 * (nTiles / NT);         // 2 M-tiles x N-groups
    for (int it = wave; it < items; it += NWAVES) {
        const int mBase = (it & 1) * 16;
        const int n0    = (it >> 1) * (NT * 16);
        const _Float16* aRow = act + (mBase + l15) * actStride + half * 16;
        const _Float16* bRow[NT];
#pragma unroll
        for (int j = 0; j < NT; ++j)
            bRow[j] = W + (size_t)(n0 + j * 16 + l15) * wStride + half * 16;

        v8f acc[NT];
#pragma unroll
        for (int j = 0; j < NT; ++j) acc[j] = (v8f){};

        // pipeline prologue: k = 0 fragments
        V16 a, b[NT];
        a.h[0] = *(const v8h*)(aRow);
        a.h[1] = *(const v8h*)(aRow + 8);
#pragma unroll
        for (int j = 0; j < NT; ++j) {
            b[j].h[0] = *(const v8h*)(bRow[j]);
            b[j].h[1] = *(const v8h*)(bRow[j] + 8);
        }
        // steady state: issue k+32 loads while k's WMMAs execute
        for (int k0 = 32; k0 < K; k0 += 32) {
            V16 an, bn[NT];
            an.h[0] = *(const v8h*)(aRow + k0);
            an.h[1] = *(const v8h*)(aRow + k0 + 8);
#pragma unroll
            for (int j = 0; j < NT; ++j) {
                bn[j].h[0] = *(const v8h*)(bRow[j] + k0);
                bn[j].h[1] = *(const v8h*)(bRow[j] + k0 + 8);
            }
#pragma unroll
            for (int j = 0; j < NT; ++j)
                acc[j] = __builtin_amdgcn_wmma_f32_16x16x32_f16(
                    false, a.v, false, b[j].v, (short)0, acc[j], false, false);
            a = an;
#pragma unroll
            for (int j = 0; j < NT; ++j) b[j] = bn[j];
        }
#pragma unroll
        for (int j = 0; j < NT; ++j)
            acc[j] = __builtin_amdgcn_wmma_f32_16x16x32_f16(
                false, a.v, false, b[j].v, (short)0, acc[j], false, false);

        // epilogue: bias (+ReLU) -> f16 LDS, or fp32 global for the final fc
#pragma unroll
        for (int j = 0; j < NT; ++j) {
            const int n = n0 + j * 16 + l15;
            if (n < nMax) {
                const float bv = bias[n];
                if (TO_LDS) {
#pragma unroll
                    for (int r = 0; r < 8; ++r) {
                        const int m = mBase + half * 8 + r;  // C layout: vgpr r
                        float v = acc[j][r] + bv;
                        if (RELU) v = fmaxf(v, 0.0f);
                        dst[m * dstStride + dstColOff + n] = (_Float16)v;
                    }
                } else {
#pragma unroll
                    for (int r = 0; r < 8; ++r) {
                        const int m = mBase + half * 8 + r;
                        gOut[(size_t)(b0 + m) * 8 + n] = acc[j][r] + bv;
                    }
                }
            }
        }
    }
}

// ---------------------------------------------------------------------------
// Fused pipeline: one workgroup handles 32 batch rows end-to-end in LDS.
// ---------------------------------------------------------------------------
__global__ __launch_bounds__(NTHREADS) void fused_kernel(
    const float* __restrict__ x,    const float* __restrict__ emb,
    const float* __restrict__ b3,   const float* __restrict__ b5,
    const float* __restrict__ b11,  const float* __restrict__ b12,
    const float* __restrict__ g1f,  const float* __restrict__ g1b,
    const float* __restrict__ g2f,  const float* __restrict__ g2b,
    const float* __restrict__ fc1b, const float* __restrict__ fc2b,
    const _Float16* __restrict__ ws, float* __restrict__ out)
{
    // LDS pool (~149 KB of 320 KB WGP budget -> 2 workgroups per WGP)
    __shared__ __align__(16) _Float16 s_pool[32 * 1024 + 32 * 768 + 32 * 512 + 32 * 64];
    __shared__ float s_bias[256];
    _Float16* const s_cat2 = s_pool;                                   // [32][1024]
    _Float16* const s_cat1 = s_pool + 32 * 1024;                       // [32][768]
    _Float16* const s_scr  = s_pool + 32 * 1024 + 32 * 768;            // [32][512]
    _Float16* const s_xa   = s_pool + 32 * 1024 + 32 * 768 + 32 * 512; // [32][64]
    _Float16* const s_h1   = s_cat1;  // fc1 out [32][1024] reuses cat1+scr (dead)

    const int tid  = threadIdx.x;
    const int lane = tid & 31;
    const int wave = __builtin_amdgcn_readfirstlane(tid) >> 5;  // SGPR -> scalar loop
    const int b0   = blockIdx.x * TILE;

    const _Float16* Wc  = ws + OFF_WC;
    const _Float16* W11 = ws + OFF_W11;
    const _Float16* W12 = ws + OFF_W12;
    const _Float16* FC1 = ws + OFF_FC1;
    const _Float16* FC2 = ws + OFF_FC2;

    // ---- stage 0: fills (all threads) + embedding lookup (wave 0) ----------
    for (int i = tid; i < 256; i += NTHREADS)
        s_bias[i] = (i < 100) ? b3[i] : ((i < 200) ? b5[i - 100] : 0.0f);
    for (int i = tid; i < 32 * 18; i += NTHREADS) {      // cat1 K-pad 750..767
        int r = i / 18, c = 750 + i % 18;
        s_cat1[r * 768 + c] = (_Float16)0.0f;
    }
    for (int i = tid; i < 32 * 24; i += NTHREADS) {      // cat2 K-pad 1000..1023
        int r = i / 24, c = 1000 + i % 24;
        s_cat2[r * 1024 + c] = (_Float16)0.0f;
    }
    if (wave == 0) {
        const float* xr = x + (size_t)(b0 + lane) * 50;
        int am = 0; float mv = xr[0];
        for (int c = 1; c < 21; ++c) { float v = xr[c]; if (v > mv) { mv = v; am = c; } }
        const float* er = emb + am * 21;
        for (int c = 0; c < 21; ++c) {
            float v = er[c];
            s_xa[lane * 64 + c]    = (_Float16)v;                // pre-ReLU (conv input)
            s_cat1[lane * 768 + c] = (_Float16)fmaxf(v, 0.0f);   // ReLU'd (GRU input)
        }
        for (int c = 21; c < 50; ++c) {
            float v = xr[c];
            s_xa[lane * 64 + c]    = (_Float16)v;
            s_cat1[lane * 768 + c] = (_Float16)fmaxf(v, 0.0f);
        }
        for (int c = 50; c < 64; ++c) s_xa[lane * 64 + c] = (_Float16)0.0f;
    }
    __syncthreads();

    // ---- GEMM1: [32x64] x Wc[256x64] -> relu(l3|l5) -> cat1 cols 50..249 ---
    gemm<4, true, true>(s_xa, 64, 64, Wc, 64, 16, lane, wave,
                        s_cat1, 768, 50, s_bias, 200, nullptr, b0);
    __syncthreads();

    // ---- GRU block 1 (T=250): Fh -> cat1[250..499], Bh -> cat1[500..749] ---
    if (wave == 0)      gru_scan(s_cat1, 768, 250, g1f, s_cat1, 768, 250, lane);
    else if (wave == 1) gru_scan(s_cat1, 768, 250, g1b, s_cat1, 768, 500, lane);
    __syncthreads();

    // ---- GEMM2: cat1[32x768] x W11 -> relu -> cat2 cols 0..499 -------------
    gemm<4, true, true>(s_cat1, 768, 768, W11, 768, 32, lane, wave,
                        s_cat2, 1024, 0, b11, 500, nullptr, b0);
    __syncthreads();

    // ---- GRU block 2 (T=500): Fh -> cat2[500..999], Bh -> scratch ----------
    if (wave == 0)      gru_scan(s_cat2, 1024, 500, g2f, s_cat2, 1024, 500, lane);
    else if (wave == 1) gru_scan(s_cat2, 1024, 500, g2b, s_scr, 512, 0, lane);
    __syncthreads();
    for (int i = tid; i < 32 * 500; i += NTHREADS) {     // o2 = Fh + Bh
        int r = i / 500, c = i % 500;
        s_cat2[r * 1024 + 500 + c] =
            (_Float16)((float)s_cat2[r * 1024 + 500 + c] + (float)s_scr[r * 512 + c]);
    }
    __syncthreads();

    // ---- GEMM3: cat2[32x1024] x W12 -> relu -> cat1 cols 0..499 ------------
    gemm<4, true, true>(s_cat2, 1024, 1024, W12, 1024, 32, lane, wave,
                        s_cat1, 768, 0, b12, 500, nullptr, b0);
    __syncthreads();

    // ---- GRU block 3 (T=500, g2 params again): x3 = Fh + Bh -> cat2[0..499]
    if (wave == 0)      gru_scan(s_cat1, 768, 500, g2f, s_cat2, 1024, 0, lane);
    else if (wave == 1) gru_scan(s_cat1, 768, 500, g2b, s_scr, 512, 0, lane);
    else {
        for (int i = tid - 64; i < 32 * 12; i += (NTHREADS - 64)) { // zero pad 500..511
            if (i >= 0) { int r = i / 12, c = 500 + i % 12; s_cat2[r * 1024 + c] = (_Float16)0.0f; }
        }
    }
    __syncthreads();
    for (int i = tid; i < 32 * 500; i += NTHREADS) {
        int r = i / 500, c = i % 500;
        s_cat2[r * 1024 + c] =
            (_Float16)((float)s_cat2[r * 1024 + c] + (float)s_scr[r * 512 + c]);
    }
    __syncthreads();

    // ---- GEMM4 (fc1): cat2[32x512] x FC1[1024x512] -> relu -> h1 -----------
    gemm<4, true, true>(s_cat2, 1024, 512, FC1, 512, 64, lane, wave,
                        s_h1, 1024, 0, fc1b, 1024, nullptr, b0);
    __syncthreads();

    // ---- GEMM5 (fc2): h1[32x1024] x FC2[16x1024] -> global out [B][8] ------
    gemm<1, false, false>(s_h1, 1024, 1024, FC2, 1024, 1, lane, wave,
                          nullptr, 0, 0, fc2b, 8, out, b0);
}

// ---------------------------------------------------------------------------
extern "C" void kernel_launch(void* const* d_in, const int* in_sizes, int n_in,
                              void* d_out, int out_size, void* d_ws, size_t ws_size,
                              hipStream_t stream) {
    const float* x    = (const float*)d_in[0];
    const float* emb  = (const float*)d_in[1];
    const float* w3   = (const float*)d_in[2];
    const float* b3   = (const float*)d_in[3];
    const float* w5   = (const float*)d_in[4];
    const float* b5   = (const float*)d_in[5];
    const float* w11  = (const float*)d_in[6];
    const float* b11  = (const float*)d_in[7];
    const float* w12  = (const float*)d_in[8];
    const float* b12  = (const float*)d_in[9];
    const float* g1f  = (const float*)d_in[10];
    const float* g1b  = (const float*)d_in[11];
    const float* g2f  = (const float*)d_in[12];
    const float* g2b  = (const float*)d_in[13];
    const float* fc1w = (const float*)d_in[14];
    const float* fc1b = (const float*)d_in[15];
    const float* fc2w = (const float*)d_in[16];
    const float* fc2b = (const float*)d_in[17];
    float* out = (float*)d_out;
    _Float16* ws = (_Float16*)d_ws;   // needs ~2.95 MB

    prep_weights<<<(WS_TOTAL + 255) / 256, 256, 0, stream>>>(
        w3, w5, w11, w12, fc1w, fc2w, ws);
    fused_kernel<<<16384 / TILE, NTHREADS, 0, stream>>>(
        x, emb, b3, b5, b11, b12, g1f, g1b, g2f, g2b, fc1b, fc2b, ws, out);
}
